// RelationAwareMLP_13271448945100
// MI455X (gfx1250) — compile-verified, matrix operand order checked by reading, not run on hardware
//
#include <hip/hip_runtime.h>

typedef __bf16 bf16;
typedef __attribute__((ext_vector_type(16))) __bf16 v16bf;
typedef __attribute__((ext_vector_type(8)))  float   v8f;

#define B_ROWS 16384
#define D_IN   512
#define D_HID  1024
#define NREL   16
#define BM 128
#define BN 128
#define BK 32
#define LDS_PAD 8
#define MAX_TILES (B_ROWS / BM + NREL)   // 144

struct TileDesc { int rowStart; int rows; int rel; };

union FragBF { v16bf v; uint4 q[2]; };

// ---------------- conversion kernels ----------------

__global__ void cvt_f32_bf16(const float* __restrict__ src, bf16* __restrict__ dst, int n) {
  int i = blockIdx.x * blockDim.x + threadIdx.x;
  int stride = gridDim.x * blockDim.x;
  for (; i < n; i += stride) dst[i] = (bf16)src[i];
}

// W is [K,N] row-major fp32 (per matrix, blockIdx.z selects matrix).
// Wt is [N,K] row-major bf16 so the GEMM's B-operand loads are K-contiguous.
__global__ __launch_bounds__(256)
void transpose_cvt(const float* __restrict__ W, bf16* __restrict__ Wt, int K, int N) {
  __shared__ float t[32][33];
  size_t matOff = (size_t)blockIdx.z * K * N;
  int n0 = blockIdx.x * 32, k0 = blockIdx.y * 32;
  int tx = threadIdx.x & 31, ty = threadIdx.x >> 5;
#pragma unroll
  for (int j = 0; j < 32; j += 8)
    t[ty + j][tx] = W[matOff + (size_t)(k0 + ty + j) * N + (n0 + tx)];
  __syncthreads();
#pragma unroll
  for (int j = 0; j < 32; j += 8)
    Wt[matOff + (size_t)(n0 + ty + j) * K + (k0 + tx)] = (bf16)t[tx][ty + j];
}

// ---------------- relation grouping (counting sort) ----------------

__global__ void zero_meta(int* meta) { if (threadIdx.x < 64) meta[threadIdx.x] = 0; }

__global__ void histo_rel(const int* __restrict__ rel, int* __restrict__ counts, int n) {
  int i = blockIdx.x * blockDim.x + threadIdx.x, s = gridDim.x * blockDim.x;
  for (; i < n; i += s) atomicAdd(&counts[rel[i]], 1);
}

__global__ void scan_build_tiles(const int* __restrict__ counts, int* __restrict__ cursor,
                                 TileDesc* __restrict__ tiles, int* __restrict__ numTiles) {
  if (threadIdx.x != 0) return;
  int running = 0, nt = 0;
  for (int r = 0; r < NREL; ++r) {
    cursor[r] = running;
    int c = counts[r];
    for (int s = 0; s < c; s += BM) {
      int rem = c - s;
      tiles[nt].rowStart = running + s;
      tiles[nt].rows = rem < BM ? rem : BM;
      tiles[nt].rel = r;
      ++nt;
    }
    running += c;
  }
  *numTiles = nt;
}

__global__ void build_perm(const int* __restrict__ rel, int* __restrict__ cursor,
                           int* __restrict__ perm, int n) {
  int i = blockIdx.x * blockDim.x + threadIdx.x, s = gridDim.x * blockDim.x;
  for (; i < n; i += s) { int p = atomicAdd(&cursor[rel[i]], 1); perm[p] = i; }
}

// ---------------- fused WMMA GEMM (+bias, +relu, gather/scatter) ----------------
// C[128x128] per block; 8 wave32s in a 4(M) x 2(N) grid; each wave: 2x4 frags of 16x16.
// Double-buffered LDS filled by GLOBAL_LOAD_ASYNC_TO_LDS_B128 (ASYNCcnt-tracked DMA).

template <bool GATHER, bool SCATTER, bool RELU, bool OUT_BF16, bool RELW>
__global__ __launch_bounds__(256)
void gemm_wmma_bf16(const bf16* __restrict__ A, const bf16* __restrict__ Wt,
                    const float* __restrict__ bias,
                    bf16* __restrict__ outB, float* __restrict__ outF,
                    const int* __restrict__ perm,
                    const TileDesc* __restrict__ tiles, const int* __restrict__ numTiles,
                    int K, int N) {
  int rowStart, rows, rel;
  if (RELW) {
    int tIdx = blockIdx.x;
    if (tIdx >= *numTiles) return;           // uniform per block
    TileDesc td = tiles[tIdx];
    rowStart = td.rowStart; rows = td.rows; rel = td.rel;
  } else {
    rowStart = blockIdx.x * BM; rows = BM; rel = 0;
  }
  const int nBase = blockIdx.y * BN;

  const bf16*  __restrict__ Wb = Wt  + (size_t)rel * N * K;   // [N,K] bf16
  const float* __restrict__ bb = bias + (size_t)rel * N;

  __shared__ __align__(16) bf16 As[2][BM][BK + LDS_PAD];
  __shared__ __align__(16) bf16 Bs[2][BN][BK + LDS_PAD];
  __shared__ __align__(16) bf16 dump[8];   // async-load sink for invalid gather rows
  __shared__ int rowDst[BM];

  const int tid = threadIdx.x;
  if (tid < BM) {
    int rl = tid;
    bool ok = rl < rows;
    int pRow = rowStart + rl;
    rowDst[rl] = SCATTER ? (ok ? perm[pRow] : 0) : pRow;
  }

  // ---- precompute per-thread staging addresses (2 A-chunks + 2 B-chunks of 16B) ----
  unsigned ldsA[2][2], ldsB[2][2];       // [assignment][buffer]
  const bf16* gA[2];
  const bf16* gB[2];
  {
    int r0 = tid >> 2;
    int c  = (tid & 3) * 8;
#pragma unroll
    for (int i = 0; i < 2; ++i) {
      int r = r0 + i * 64;
      int src = (r < rows) ? (GATHER ? perm[rowStart + r] : rowStart + r) : -1;
      if (src >= 0) {
        gA[i] = A + (size_t)src * K + c;
        ldsA[i][0] = (unsigned)(size_t)&As[0][r][c];
        ldsA[i][1] = (unsigned)(size_t)&As[1][r][c];
      } else {
        // keep async issue non-divergent: read a valid dummy row, land in dump,
        // and pre-zero the real LDS destination once (never overwritten later).
        gA[i] = A + c;
        unsigned d = (unsigned)(size_t)&dump[0];
        ldsA[i][0] = d; ldsA[i][1] = d;
        uint4 z = {0u, 0u, 0u, 0u};
        *(uint4*)&As[0][r][c] = z;
        *(uint4*)&As[1][r][c] = z;
      }
      gB[i] = Wb + (size_t)(nBase + r) * K + c;
      ldsB[i][0] = (unsigned)(size_t)&Bs[0][r][c];
      ldsB[i][1] = (unsigned)(size_t)&Bs[1][r][c];
    }
  }

  auto issueStage = [&](int k0, int buf) {   // 4 async-to-LDS ops per wave per stage
#pragma unroll
    for (int i = 0; i < 2; ++i) {
      unsigned long long ga = (unsigned long long)(size_t)(gA[i] + k0);
      asm volatile("global_load_async_to_lds_b128 %0, %1, off"
                   :: "v"(ldsA[i][buf]), "v"(ga) : "memory");
      unsigned long long gb = (unsigned long long)(size_t)(gB[i] + k0);
      asm volatile("global_load_async_to_lds_b128 %0, %1, off"
                   :: "v"(ldsB[i][buf]), "v"(gb) : "memory");
    }
  };

  const int lane = tid & 31;
  const int laneLo = lane & 15;
  const int laneHi = lane >> 4;
  const int waveId = tid >> 5;
  const int wm = (waveId & 3) * 32;   // wave M offset within tile
  const int wn = (waveId >> 2) * 64;  // wave N offset within tile

  v8f acc[2][4];
  const v8f vzero = {0.f, 0.f, 0.f, 0.f, 0.f, 0.f, 0.f, 0.f};
#pragma unroll
  for (int mi = 0; mi < 2; ++mi)
#pragma unroll
    for (int ni = 0; ni < 4; ++ni) acc[mi][ni] = vzero;

  __syncthreads();              // pre-zeros + rowDst in place
  const int nSteps = K / BK;
  issueStage(0, 0);

  for (int s = 0; s < nSteps; ++s) {
    const int buf = s & 1;
    if (s + 1 < nSteps) {
      issueStage((s + 1) * BK, buf ^ 1);   // stream next stage into other buffer
      // in-order completion: <=4 outstanding => stage s fully landed in LDS
      asm volatile("s_wait_asynccnt 0x4" ::: "memory");
    } else {
      asm volatile("s_wait_asynccnt 0x0" ::: "memory");
    }
    __syncthreads();            // stage s visible to all waves

    const bf16 (*Asb)[BK + LDS_PAD] = As[buf];
    const bf16 (*Bsb)[BK + LDS_PAD] = Bs[buf];

    // Operand fragments per documented 16-bit layout:
    // lane<16: K 0..7 and 16..23 ; lane>=16: K 8..15 and 24..31.
    FragBF a[2], b[4];
#pragma unroll
    for (int mi = 0; mi < 2; ++mi) {
      int rr = wm + mi * 16 + laneLo;
      a[mi].q[0] = *(const uint4*)&Asb[rr][8 * laneHi];
      a[mi].q[1] = *(const uint4*)&Asb[rr][16 + 8 * laneHi];
    }
#pragma unroll
    for (int ni = 0; ni < 4; ++ni) {
      int cc = wn + ni * 16 + laneLo;
      b[ni].q[0] = *(const uint4*)&Bsb[cc][8 * laneHi];
      b[ni].q[1] = *(const uint4*)&Bsb[cc][16 + 8 * laneHi];
    }
#pragma unroll
    for (int mi = 0; mi < 2; ++mi)
#pragma unroll
      for (int ni = 0; ni < 4; ++ni)
        acc[mi][ni] = __builtin_amdgcn_wmma_f32_16x16x32_bf16(
            false, a[mi].v, false, b[ni].v, (short)0, acc[mi][ni], false, false);

    __syncthreads();            // all reads of buf done before it is refilled
  }

  // Epilogue: bias (+relu), store bf16 (intermediate) or scattered f32 (final).
#pragma unroll
  for (int mi = 0; mi < 2; ++mi) {
#pragma unroll
    for (int ni = 0; ni < 4; ++ni) {
      int nG = nBase + wn + ni * 16 + laneLo;
      float bv = bb[nG];
#pragma unroll
      for (int v = 0; v < 8; ++v) {
        int rl = wm + mi * 16 + laneHi * 8 + v;   // C layout: M = 8*laneHi + vgpr
        float val = acc[mi][ni][v] + bv;
        if (RELU) val = fmaxf(val, 0.0f);
        if (rl < rows) {
          int dst = rowDst[rl];
          if (OUT_BF16) outB[(size_t)dst * N + nG] = (bf16)val;
          else          outF[(size_t)dst * N + nG] = val;
        }
      }
    }
  }
}

// ---------------- host launch ----------------

extern "C" void kernel_launch(void* const* d_in, const int* in_sizes, int n_in,
                              void* d_out, int out_size, void* d_ws, size_t ws_size,
                              hipStream_t stream) {
  (void)in_sizes; (void)n_in; (void)out_size; (void)ws_size;
  const float* x   = (const float*)d_in[0];
  const int*   rel = (const int*)  d_in[1];
  const float* W0  = (const float*)d_in[2];
  const float* b0  = (const float*)d_in[3];
  const float* W1  = (const float*)d_in[4];
  const float* b1  = (const float*)d_in[5];
  const float* RW1 = (const float*)d_in[6];
  const float* Rb1 = (const float*)d_in[7];
  const float* RW2 = (const float*)d_in[8];
  const float* Rb2 = (const float*)d_in[9];
  const float* RW3 = (const float*)d_in[10];
  const float* Rb3 = (const float*)d_in[11];
  float* out = (float*)d_out;

  char* p = (char*)d_ws;
  auto take = [&](size_t bytes) -> char* {
    char* r = p;
    p += (bytes + 255) & ~(size_t)255;
    return r;
  };
  bf16* xb  = (bf16*)take((size_t)B_ROWS * D_IN  * 2);
  bf16* hA  = (bf16*)take((size_t)B_ROWS * D_HID * 2);
  bf16* hB  = (bf16*)take((size_t)B_ROWS * D_HID * 2);
  bf16* W0t = (bf16*)take((size_t)D_IN  * D_HID * 2);
  bf16* W1t = (bf16*)take((size_t)D_HID * D_HID * 2);
  bf16* R1t = (bf16*)take((size_t)NREL * D_HID * D_HID * 2);
  bf16* R2t = (bf16*)take((size_t)NREL * D_HID * D_HID * 2);
  bf16* R3t = (bf16*)take((size_t)NREL * D_HID * D_HID * 2);
  int*  perm = (int*)take((size_t)B_ROWS * 4);
  int*  meta = (int*)take(64 * 4);
  TileDesc* tiles = (TileDesc*)take(MAX_TILES * sizeof(TileDesc));
  int* counts = meta;        // [16]
  int* cursor = meta + 16;   // [16]
  int* numTiles = meta + 32; // [1]

  // 1) one-time bf16 conversion (weights transposed to [N,K])
  cvt_f32_bf16<<<2048, 256, 0, stream>>>(x, xb, B_ROWS * D_IN);
  transpose_cvt<<<dim3(D_HID / 32, D_IN / 32, 1),     256, 0, stream>>>(W0,  W0t, D_IN,  D_HID);
  transpose_cvt<<<dim3(D_HID / 32, D_HID / 32, 1),    256, 0, stream>>>(W1,  W1t, D_HID, D_HID);
  transpose_cvt<<<dim3(D_HID / 32, D_HID / 32, NREL), 256, 0, stream>>>(RW1, R1t, D_HID, D_HID);
  transpose_cvt<<<dim3(D_HID / 32, D_HID / 32, NREL), 256, 0, stream>>>(RW2, R2t, D_HID, D_HID);
  transpose_cvt<<<dim3(D_HID / 32, D_HID / 32, NREL), 256, 0, stream>>>(RW3, R3t, D_HID, D_HID);

  // 2) group rows by relation (counting sort) + tile worklist
  zero_meta<<<1, 64, 0, stream>>>(meta);
  histo_rel<<<128, 256, 0, stream>>>(rel, counts, B_ROWS);
  scan_build_tiles<<<1, 32, 0, stream>>>(counts, cursor, tiles, numTiles);
  build_perm<<<128, 256, 0, stream>>>(rel, cursor, perm, B_ROWS);

  dim3 gFull(B_ROWS / BM, D_HID / BN);   // 128 x 8
  dim3 gRel(MAX_TILES,    D_HID / BN);   // 144 x 8 (extra blocks exit on numTiles)

  // 3) dense stack
  gemm_wmma_bf16<false, false, true, true, false><<<gFull, 256, 0, stream>>>(
      xb, W0t, b0, hA, nullptr, perm, tiles, numTiles, D_IN, D_HID);
  gemm_wmma_bf16<false, false, true, true, false><<<gFull, 256, 0, stream>>>(
      hA, W1t, b1, hB, nullptr, perm, tiles, numTiles, D_HID, D_HID);

  // 4) relation layers as grouped GEMMs: gather -> (permuted) -> scatter
  gemm_wmma_bf16<true,  false, true,  true,  true><<<gRel, 256, 0, stream>>>(
      hB, R1t, Rb1, hA, nullptr, perm, tiles, numTiles, D_HID, D_HID);
  gemm_wmma_bf16<false, false, true,  true,  true><<<gRel, 256, 0, stream>>>(
      hA, R2t, Rb2, hB, nullptr, perm, tiles, numTiles, D_HID, D_HID);
  gemm_wmma_bf16<false, true,  false, false, true><<<gRel, 256, 0, stream>>>(
      hB, R3t, Rb3, nullptr, out, perm, tiles, numTiles, D_HID, D_HID);
}
